// Model_50508815401654
// MI455X (gfx1250) — matrix operand order repaired and checked by reading, not compiled, not run on hardware
//
#include <hip/hip_runtime.h>
#include <hip/hip_bf16.h>

// ---------------- problem constants (from reference) ----------------
#define NNODES   64000
#define NPG      1000
#define NGRAPH   64
#define FDIM     128
#define NHEAD    4
#define DDIM     32
#define TD       128        // NHEAD * DDIM
#define NEDGE    2048000
#define TOPK     30
#define C1CH     16
#define C2CH     32
#define KW2      5
#define P1LEN    15         // 30/2
#define C2LEN    11         // 15-5+1
#define OUTPG    (C2CH * C2LEN)   // 352

typedef __attribute__((ext_vector_type(16))) _Float16 v16h;
typedef __attribute__((ext_vector_type(8)))  float    v8f;

// ---------------- WhT prep: W (H,F,D) f32  ->  WhT[n][k] f16, n = h*32+d ----
__global__ void prep_wht(const float* __restrict__ W, _Float16* __restrict__ WhT) {
    int idx = blockIdx.x * blockDim.x + threadIdx.x;   // 128*128 elements
    if (idx >= TD * FDIM) return;
    int n = idx >> 7;          // output column
    int k = idx & 127;         // input feature
    int h = n >> 5;
    int d = n & 31;
    WhT[idx] = (_Float16)W[(h * FDIM + k) * DDIM + d];
}

// ---------------- GEMM: hfeat (N,128) = node_feat (N,128) x Wc (128,128) ----
// 128 threads = 4 waves; each wave computes a 16-row strip x all 128 cols.
__device__ inline v16h cvt_a_frag(float4 a0, float4 a1v, float4 a2v, float4 a3v) {
    v16h r;
    r[0]=(_Float16)a0.x;  r[1]=(_Float16)a0.y;  r[2]=(_Float16)a0.z;  r[3]=(_Float16)a0.w;
    r[4]=(_Float16)a1v.x; r[5]=(_Float16)a1v.y; r[6]=(_Float16)a1v.z; r[7]=(_Float16)a1v.w;
    r[8]=(_Float16)a2v.x; r[9]=(_Float16)a2v.y; r[10]=(_Float16)a2v.z; r[11]=(_Float16)a2v.w;
    r[12]=(_Float16)a3v.x; r[13]=(_Float16)a3v.y; r[14]=(_Float16)a3v.z; r[15]=(_Float16)a3v.w;
    return r;
}

__global__ void __launch_bounds__(128)
gemm_wmma(const float* __restrict__ x, const _Float16* __restrict__ WhT,
          float* __restrict__ hfeat) {
    const int wave  = threadIdx.x >> 5;
    const int lane  = threadIdx.x & 31;
    const int nlow  = lane & 15;
    const int mbase = blockIdx.x * 64 + wave * 16;
    const int mrow  = mbase + nlow;                 // A row for this lane
    const int hiA   = (lane & 16) ? 16 : 0;         // A-frag K-half select
    const int hiB   = (lane & 16) ? 16 : 0;         // B-frag K-half select

    v8f acc[8];
#pragma unroll
    for (int nt = 0; nt < 8; ++nt) acc[nt] = v8f{};

    const float* arowbase = x + (size_t)mrow * FDIM;
#pragma unroll
    for (int k0 = 0; k0 < FDIM; k0 += 32) {
        // A fragment: lanes 0-15 hold K k0+0..15; lanes 16-31 hold K k0+16..31
        const float* ap = arowbase + k0 + hiA;
        float4 a0  = *(const float4*)(ap);
        float4 a1v = *(const float4*)(ap + 4);
        float4 a2v = *(const float4*)(ap + 8);
        float4 a3v = *(const float4*)(ap + 12);
        v16h af = cvt_a_frag(a0, a1v, a2v, a3v);
#pragma unroll
        for (int nt = 0; nt < 8; ++nt) {
            // B fragment: lane holds col n = nt*16 + nlow, 16 contiguous K halves
            const _Float16* bp = WhT + (size_t)(nt * 16 + nlow) * FDIM + k0 + hiB;
            v16h bf = *(const v16h*)bp;
            acc[nt] = __builtin_amdgcn_wmma_f32_16x16x32_f16(
                false, af, false, bf, (short)0, acc[nt], false, false);
        }
    }
    // C/D layout: VGPR r -> M = r (+8 for lanes 16-31), N = lane&15
    const int mstore = mbase + ((lane & 16) ? 8 : 0);
#pragma unroll
    for (int nt = 0; nt < 8; ++nt) {
#pragma unroll
        for (int r = 0; r < 8; ++r) {
            hfeat[(size_t)(mstore + r) * TD + nt * 16 + nlow] = acc[nt][r];
        }
    }
}

// ---------------- attention scalars f1 = h.a1, f2 = h.a2 -------------------
__global__ void attn_scalar(const float* __restrict__ hfeat,
                            const float* __restrict__ a1, const float* __restrict__ a2,
                            float* __restrict__ f1, float* __restrict__ f2) {
    int idx = blockIdx.x * blockDim.x + threadIdx.x;   // N * H
    if (idx >= NNODES * NHEAD) return;
    int n = idx >> 2, h = idx & 3;
    const float* hp = hfeat + (size_t)n * TD + h * DDIM;
    const float* p1 = a1 + h * DDIM;
    const float* p2 = a2 + h * DDIM;
    float s1 = 0.f, s2 = 0.f;
#pragma unroll
    for (int d = 0; d < DDIM; ++d) { float v = hp[d]; s1 += v * p1[d]; s2 += v * p2[d]; }
    f1[idx] = s1; f2[idx] = s2;
}

// ---------------- zero fill -------------------------------------------------
__global__ void zero_f32(float* __restrict__ p, long n) {
    long i = (long)blockIdx.x * blockDim.x + threadIdx.x;
    long stride = (long)gridDim.x * blockDim.x;
    for (; i < n; i += stride) p[i] = 0.f;
}

// ---------------- edge scatter: one wave per edge, one lane per dim --------
__global__ void edge_kernel(const long long* __restrict__ src,
                            const long long* __restrict__ dst,
                            const float* __restrict__ f1, const float* __restrict__ f2,
                            const float* __restrict__ hfeat,
                            float* __restrict__ msg, float* __restrict__ rowsum) {
    const int lane   = threadIdx.x & 31;
    long wid         = ((long)blockIdx.x * blockDim.x + threadIdx.x) >> 5;
    const long nwave = ((long)gridDim.x * blockDim.x) >> 5;
    for (long e = wid; e < NEDGE; e += nwave) {
        const long s = (long)src[e];
        const long d = (long)dst[e];
        float w[NHEAD];
#pragma unroll
        for (int h = 0; h < NHEAD; ++h) {
            float v = f1[s * NHEAD + h] + f2[d * NHEAD + h];
            float l = v > 0.f ? v : 0.2f * v;           // leaky_relu(0.2)
            w[h] = __expf(-l);
        }
        const float* hd = hfeat + d * TD;
        float* mp = msg + s * TD;
#pragma unroll
        for (int h = 0; h < NHEAD; ++h) {
            float hv = hd[h * DDIM + lane];
            unsafeAtomicAdd(mp + h * DDIM + lane, w[h] * hv);
        }
        if (lane < NHEAD) unsafeAtomicAdd(rowsum + s * NHEAD + lane, w[lane]);
    }
}

// ---------------- normalize + ELU (in place: msg -> cur) -------------------
__global__ void elu_norm(float* __restrict__ msg, const float* __restrict__ rowsum) {
    long i = (long)blockIdx.x * blockDim.x + threadIdx.x;
    if (i >= (long)NNODES * TD) return;
    long n = i >> 7;
    int  c = (int)(i & 127);
    float r = rowsum[n * NHEAD + (c >> 5)];
    float v = msg[i] / r;
    msg[i] = v > 0.f ? v : (__expf(v) - 1.f);           // elu
}

// ---------------- top-K=30 per graph on channel 127 ------------------------
__global__ void __launch_bounds__(128)
topk_kernel(const float* __restrict__ cur, int* __restrict__ topk_idx) {
    __shared__ float keys[NPG];
    __shared__ float rv[128];
    __shared__ int   ri[128];
    const int b = blockIdx.x, t = threadIdx.x;
    for (int i = t; i < NPG; i += 128)
        keys[i] = cur[((size_t)b * NPG + i) * TD + (TD - 1)];
    __syncthreads();
    for (int k = 0; k < TOPK; ++k) {
        float best = -INFINITY; int bi = NPG;
        for (int i = t; i < NPG; i += 128) {
            float v = keys[i];
            if (v > best) { best = v; bi = i; }          // first (lowest) index wins ties
        }
        rv[t] = best; ri[t] = bi;
        __syncthreads();
        for (int s = 64; s > 0; s >>= 1) {
            if (t < s) {
                if (rv[t + s] > rv[t] || (rv[t + s] == rv[t] && ri[t + s] < ri[t])) {
                    rv[t] = rv[t + s]; ri[t] = ri[t + s];
                }
            }
            __syncthreads();
        }
        if (t == 0) { topk_idx[b * TOPK + k] = ri[0]; keys[ri[0]] = -INFINITY; }
        __syncthreads();
    }
}

// ---------------- conv head: conv1+relu -> maxpool2 -> conv2+relu ----------
__global__ void __launch_bounds__(128)
conv_head(const float* __restrict__ cur, const int* __restrict__ topk_idx,
          const float* __restrict__ w1, const float* __restrict__ b1,
          const float* __restrict__ w2, const float* __restrict__ b2,
          float* __restrict__ out) {
    __shared__ float pooled[TOPK][TD];
    __shared__ float c1s[C1CH][TOPK];
    __shared__ float p1s[C1CH][P1LEN + 1];
    const int b = blockIdx.x, t = threadIdx.x;

    for (int i = t; i < TOPK * TD; i += 128) {
        int k = i >> 7, c = i & 127;
        int node = topk_idx[b * TOPK + k];
        pooled[k][c] = cur[((size_t)b * NPG + node) * TD + c];
    }
    __syncthreads();
    // conv1: kernel width TD, stride TD => plain dot over each pooled row
    for (int i = t; i < C1CH * TOPK; i += 128) {
        int c = i / TOPK, k = i % TOPK;
        float acc = b1[c];
        const float* wr = w1 + c * TD;
#pragma unroll 16
        for (int j = 0; j < TD; ++j) acc += pooled[k][j] * wr[j];
        c1s[c][k] = fmaxf(acc, 0.f);
    }
    __syncthreads();
    for (int i = t; i < C1CH * P1LEN; i += 128) {
        int c = i / P1LEN, j = i % P1LEN;
        p1s[c][j] = fmaxf(c1s[c][2 * j], c1s[c][2 * j + 1]);
    }
    __syncthreads();
    for (int i = t; i < OUTPG; i += 128) {
        int o = i / C2LEN, tt = i % C2LEN;
        float acc = b2[o];
#pragma unroll
        for (int c = 0; c < C1CH; ++c)
#pragma unroll
            for (int w = 0; w < KW2; ++w)
                acc += p1s[c][tt + w] * w2[(o * C1CH + c) * KW2 + w];
        out[(size_t)b * OUTPG + i] = fmaxf(acc, 0.f);   // relu (outer relu is idempotent)
    }
}

// ---------------- launcher --------------------------------------------------
extern "C" void kernel_launch(void* const* d_in, const int* in_sizes, int n_in,
                              void* d_out, int out_size, void* d_ws, size_t ws_size,
                              hipStream_t stream) {
    const float*     node_feat = (const float*)d_in[0];
    const float*     W         = (const float*)d_in[1];
    const float*     a1        = (const float*)d_in[2];
    const float*     a2        = (const float*)d_in[3];
    const float*     w1        = (const float*)d_in[4];
    const float*     b1        = (const float*)d_in[5];
    const float*     w2        = (const float*)d_in[6];
    const float*     b2        = (const float*)d_in[7];
    const long long* eidx      = (const long long*)d_in[8];   // int64 (2, E)
    float*           out       = (float*)d_out;

    // workspace carve-out
    char* ws = (char*)d_ws;
    size_t off = 0;
    auto carve = [&](size_t bytes) -> void* {
        void* p = ws + off;
        off += (bytes + 255) & ~(size_t)255;
        return p;
    };
    float*     hfeat  = (float*)    carve((size_t)NNODES * TD * sizeof(float));
    float*     msg    = (float*)    carve((size_t)NNODES * TD * sizeof(float));
    _Float16*  WhT    = (_Float16*) carve((size_t)TD * FDIM * sizeof(_Float16));
    float*     f1     = (float*)    carve((size_t)NNODES * NHEAD * sizeof(float));
    float*     f2     = (float*)    carve((size_t)NNODES * NHEAD * sizeof(float));
    float*     rowsum = (float*)    carve((size_t)NNODES * NHEAD * sizeof(float));
    int*       tkidx  = (int*)      carve((size_t)NGRAPH * TOPK * sizeof(int));

    const long long* src = eidx;             // edge_index[0]
    const long long* dst = eidx + NEDGE;     // edge_index[1]

    prep_wht<<<(TD * FDIM + 255) / 256, 256, 0, stream>>>(W, WhT);
    zero_f32<<<2048, 256, 0, stream>>>(msg, (long)NNODES * TD);
    zero_f32<<<256, 256, 0, stream>>>(rowsum, (long)NNODES * NHEAD);

    gemm_wmma<<<NNODES / 64, 128, 0, stream>>>(node_feat, WhT, hfeat);

    attn_scalar<<<(NNODES * NHEAD + 255) / 256, 256, 0, stream>>>(hfeat, a1, a2, f1, f2);

    edge_kernel<<<2048, 256, 0, stream>>>(src, dst, f1, f2, hfeat, msg, rowsum);

    elu_norm<<<(NNODES * TD + 255) / 256, 256, 0, stream>>>(msg, rowsum);

    topk_kernel<<<NGRAPH, 128, 0, stream>>>(msg, tkidx);

    conv_head<<<NGRAPH, 128, 0, stream>>>(msg, tkidx, w1, b1, w2, b2, out);
}
